// CapsNet_43525198577830
// MI455X (gfx1250) — compile-verified, matrix operand order checked by reading, not run on hardware
//
#include <hip/hip_runtime.h>

typedef __bf16 bf16_t;
typedef __attribute__((ext_vector_type(16))) __bf16 v16bf;
typedef __attribute__((ext_vector_type(8)))  float  v8f;

#define BM 128
#define BN 128
#define BK 32
#define LDS_PITCH 40   // bf16 elems per LDS row: 80B (16B aligned), conflict-free gathers
#define SPLITK 3       // prim-conv K split: 20736 = 3 * 6912 (6912 = 27 patches * 256ch)

union FragU { unsigned int u[8]; v16bf v; };

__device__ __forceinline__ bf16_t f2bf(float f) {
    unsigned u = __builtin_bit_cast(unsigned, f);
    unsigned r = u + 0x7FFFu + ((u >> 16) & 1u);   // round-to-nearest-even
    unsigned short h = (unsigned short)(r >> 16);
    return __builtin_bit_cast(bf16_t, h);
}

// Gather one 16x32 bf16 WMMA fragment from an LDS tile (rows x 32k, pitch LDS_PITCH).
// Layout per CDNA5 ISA 7.12.2 (16-bit A / B 32x16 mirrored): lane l -> row (l&15),
// VGPR v -> K = (v>>2)*16 + (l>>4)*8 + (v&3)*2, packed {k, k+1} in one dword.
// Per lane this is two contiguous 16B runs -> compiler emits ds_load_b128 pairs.
__device__ __forceinline__ v16bf load_frag(const bf16_t* base, int row0, int lane) {
    const int r = row0 + (lane & 15);
    const int khalf = (lane >> 4) << 3;
    const unsigned int* p = (const unsigned int*)(base + r * LDS_PITCH);
    FragU f;
#pragma unroll
    for (int v = 0; v < 8; ++v) {
        const int k = ((v >> 2) << 4) + khalf + ((v & 3) << 1);
        f.u[v] = p[k >> 1];
    }
    return f.v;
}

// Generic bf16 WMMA GEMM: C[z][m][n] = sum_k A[z][m][k] * B[z][n][k]
// AMODE 0: A row-major (lda).  AMODE 1: implicit im2col of NHWC conv1-out for the
//          stride-2 9x9 primary-caps conv (m -> (nimg, oh, ow); k -> (kh,kw,c)).
// z axis serves either batch (aZ/bZ/cZ offsets) or split-K (kOffZ elements per z).
// Double-buffered LDS: global fetch of tile kb+1 overlaps WMMA on tile kb.
template<int AMODE, bool NCHK, bool HASBIAS, bool RELU, bool OUTBF16>
__global__ __launch_bounds__(256) void gemm_wmma(
    const bf16_t* __restrict__ A, const bf16_t* __restrict__ B,
    void* __restrict__ C, const float* __restrict__ bias,
    int Nn, int K, int lda, int ldb, int ldc,
    long aZ, long bZ, long cZ, long kOffZ)
{
    __shared__ bf16_t sA[2][BM * LDS_PITCH];
    __shared__ bf16_t sB[2][BN * LDS_PITCH];
    const int t    = threadIdx.x;
    const int lane = t & 31;
    const int wid  = t >> 5;
    const int z    = blockIdx.z;
    const long M0  = (long)blockIdx.x * BM;
    const int  N0  = blockIdx.y * BN;
    const bf16_t* Ab = A + z * aZ;
    const bf16_t* Bb = B + z * bZ;
    const long kz  = (long)z * kOffZ;
    const int waveM = (wid >> 2) * 64;   // 2 wave-rows
    const int waveN = (wid & 3) * 32;    // 4 wave-cols

    const int row = t >> 1;          // 0..127: tile row loaded by this thread
    const int kh  = (t & 1) << 4;    // 0 or 16: which 16-elem half of the K slice
    const int ksteps = K / BK;

    // per-thread loop-invariant addressing
    long aRow0;
    if (AMODE == 0) {
        aRow0 = (M0 + row) * (long)lda;
    } else {
        const long m  = M0 + row;
        const int nimg = (int)(m >> 6);
        const int oh   = ((int)m >> 3) & 7;
        const int ow   = (int)m & 7;
        aRow0 = ((long)nimg * 24 + oh * 2) * 24 + ow * 2;   // NHWC row base (x256 later)
    }
    const int  nglob  = N0 + row;
    const bool bvalid = !NCHK || (nglob < Nn);
    const long bRow0  = (long)nglob * ldb;

    uint4 a0, a1, b0, b1;
    auto fetch = [&](int kb) {
        const long kbase = kz + (long)kb * BK + kh;
        if (AMODE == 0) {
            const bf16_t* src = Ab + aRow0 + kbase;
            a0 = *(const uint4*)src;
            a1 = *(const uint4*)(src + 8);
        } else {
            const int kk    = (int)kbase;
            const int patch = kk >> 8;          // (kh*9+kw): constant over 32-chunk
            const int c     = kk & 255;
            const int khh = patch / 9;
            const int kww = patch - khh * 9;
            const bf16_t* src = Ab + ((aRow0 + khh * 24 + kww) << 8) + c;
            a0 = *(const uint4*)src;
            a1 = *(const uint4*)(src + 8);
        }
        if (bvalid) {
            const bf16_t* src = Bb + bRow0 + kbase;
            b0 = *(const uint4*)src;
            b1 = *(const uint4*)(src + 8);
        } else {
            b0 = make_uint4(0,0,0,0);
            b1 = make_uint4(0,0,0,0);
        }
    };

    v8f acc[4][2];
#pragma unroll
    for (int i = 0; i < 4; ++i)
#pragma unroll
        for (int j = 0; j < 2; ++j)
#pragma unroll
            for (int e = 0; e < 8; ++e) acc[i][j][e] = 0.0f;

    fetch(0);
    int buf = 0;
    for (int kb = 0; kb < ksteps; ++kb) {
        // commit staged tile (waits only on this thread's outstanding loads)
        *(uint4*)(sA[buf] + row * LDS_PITCH + kh)     = a0;
        *(uint4*)(sA[buf] + row * LDS_PITCH + kh + 8) = a1;
        *(uint4*)(sB[buf] + row * LDS_PITCH + kh)     = b0;
        *(uint4*)(sB[buf] + row * LDS_PITCH + kh + 8) = b1;
        __syncthreads();
        // issue next tile's global loads; they complete during the WMMA phase
        if (kb + 1 < ksteps) fetch(kb + 1);

        v16bf bf0 = load_frag(sB[buf], waveN,      lane);
        v16bf bf1 = load_frag(sB[buf], waveN + 16, lane);
#pragma unroll
        for (int mt = 0; mt < 4; ++mt) {
            v16bf af = load_frag(sA[buf], waveM + mt * 16, lane);
            acc[mt][0] = __builtin_amdgcn_wmma_f32_16x16x32_bf16(
                false, af, false, bf0, (short)0, acc[mt][0], false, false);
            acc[mt][1] = __builtin_amdgcn_wmma_f32_16x16x32_bf16(
                false, af, false, bf1, (short)0, acc[mt][1], false, false);
        }
        buf ^= 1;
    }

    // epilogue: C/D layout -> lane holds col (lane&15), VGPR v holds row v + 8*(lane>>4)
    const int halfm = (lane >> 4) << 3;
#pragma unroll
    for (int mt = 0; mt < 4; ++mt) {
#pragma unroll
        for (int nt = 0; nt < 2; ++nt) {
            const int n = N0 + waveN + nt * 16 + (lane & 15);
            if (NCHK && n >= Nn) continue;
            const float bv = HASBIAS ? bias[n] : 0.0f;
#pragma unroll
            for (int v = 0; v < 8; ++v) {
                const long m = M0 + waveM + mt * 16 + v + halfm;
                float val = acc[mt][nt][v] + bv;
                if (RELU) val = fmaxf(val, 0.0f);
                const long off = (long)z * cZ + m * (long)ldc + n;
                if (OUTBF16) ((bf16_t*)C)[off] = f2bf(val);
                else         ((float*)C)[off]  = val;
            }
        }
    }
}

// sum SPLITK partials + bias -> xprim (16384 x 256)
__global__ void k_reduce(const float* __restrict__ part, const float* __restrict__ bias,
                         float* __restrict__ out)
{
    const long idx = (long)blockIdx.x * 256 + threadIdx.x;
    const int  n   = (int)(idx & 255);
    float s = bias[n];
#pragma unroll
    for (int zz = 0; zz < SPLITK; ++zz) s += part[(long)zz * (16384L * 256) + idx];
    out[idx] = s;
}

// conv1 weights (256,3,9,9) -> bf16 [o][k], k = c*81 + kh*9 + kw, padded to 256
__global__ void k_wc1(const float* __restrict__ w, bf16_t* __restrict__ W) {
    const int o = blockIdx.x, k = threadIdx.x;
    float val = 0.0f;
    if (k < 243) { const int c = k / 81, r = k - c * 81; val = w[((long)o * 3 + c) * 81 + r]; }
    W[(long)o * 256 + k] = f2bf(val);
}

// im2col for conv1: A1[m][k] bf16, m=(nimg,oh,ow) over 24x24, K padded 243->256
__global__ void k_im2col1(const float* __restrict__ in, bf16_t* __restrict__ A1) {
    const int m = blockIdx.x, k = threadIdx.x;
    const int ow = m % 24, oh = (m / 24) % 24, nimg = m / 576;
    float val = 0.0f;
    if (k < 243) {
        const int c = k / 81, rr = k - c * 81, khh = rr / 9, kww = rr - khh * 9;
        val = in[(((long)nimg * 3 + c) * 32 + (oh + khh)) * 32 + (ow + kww)];
    }
    A1[(long)m * 256 + k] = f2bf(val);
}

// prim weights (256,256,9,9) -> bf16 [o][k], k = (kh*9+kw)*256 + cin (matches NHWC act)
__global__ void k_wp(const float* __restrict__ w, bf16_t* __restrict__ W) {
    const int patch = blockIdx.x;   // 0..80
    const int o     = blockIdx.y;   // 0..255
    const int c     = threadIdx.x;  // 0..255
    W[(long)o * 20736 + patch * 256 + c] = f2bf(w[((long)o * 256 + c) * 81 + patch]);
}

// caps_w (64,256,160) -> bf16 [i][n][k] (k = input dim, contiguous)
__global__ void k_wcaps(const float* __restrict__ w, bf16_t* __restrict__ W) {
    const int nn = blockIdx.x;   // 0..159
    const int i  = blockIdx.y;   // 0..63
    const int k  = threadIdx.x;  // 0..255
    W[((long)i * 160 + nn) * 256 + k] = f2bf(w[((long)i * 256 + k) * 160 + nn]);
}

// squash over dim=256 per (nimg, caps) row; one wave per row; fp32 in -> bf16 out
__global__ void k_squash(const float* __restrict__ xin, bf16_t* __restrict__ xout) {
    const int row  = blockIdx.x * 8 + (threadIdx.x >> 5);
    const int lane = threadIdx.x & 31;
    const float* r = xin + (long)row * 256;
    float v[8], l2 = 0.0f;
#pragma unroll
    for (int q = 0; q < 8; ++q) { v[q] = r[lane + q * 32]; l2 += v[q] * v[q]; }
#pragma unroll
    for (int off = 16; off > 0; off >>= 1) l2 += __shfl_xor(l2, off, 32);
    const float sc = l2 / (1.0f + l2) * __frsqrt_rn(l2);
    bf16_t* o = xout + (long)row * 256;
#pragma unroll
    for (int q = 0; q < 8; ++q) o[lane + q * 32] = f2bf(v[q] * sc);
}

// dynamic routing: one workgroup per sample; u tile (64x10x16) lives in LDS
__global__ void k_routing(const float* __restrict__ u, const float* __restrict__ b_logits,
                          float* __restrict__ probs_out, float* __restrict__ onehot)
{
    __shared__ float su[64 * 160];
    __shared__ float sb[640];
    __shared__ float sc[640];
    __shared__ float ss[160];
    __shared__ float sv[160];
    __shared__ float sc0[10];
    __shared__ float sscale[10];
    __shared__ int   slabel;
    const int n = blockIdx.x, t = threadIdx.x;
    const float* un = u + (long)n * 10240;
    for (int i = t; i < 10240; i += 256) su[i] = un[i];
    if (t < 10) {
        float mx = b_logits[0];
        for (int j = 1; j < 10; ++j) mx = fmaxf(mx, b_logits[j]);
        float sum = 0.0f;
        for (int j = 0; j < 10; ++j) sum += __expf(b_logits[j] - mx);
        sc0[t] = __expf(b_logits[t] - mx) / sum;
    }
    for (int i = t; i < 640; i += 256) sb[i] = b_logits[i % 10];
    __syncthreads();
    if (t < 160) {
        float s = 0.0f;
        for (int i = 0; i < 64; ++i) s += su[i * 160 + t];
        ss[t] = s * sc0[t >> 4];
    }
    __syncthreads();
    if (t < 10) {
        float l2 = 0.0f;
        for (int d = 0; d < 16; ++d) { const float x = ss[t * 16 + d]; l2 += x * x; }
        sscale[t] = l2 / (1.0f + l2) * __frsqrt_rn(l2);
    }
    __syncthreads();
    if (t < 160) sv[t] = ss[t] * sscale[t >> 4];
    __syncthreads();
    for (int it = 0; it < 3; ++it) {
        for (int idx = t; idx < 640; idx += 256) {
            const int i = idx / 10, j = idx - i * 10;
            float d = 0.0f;
            for (int q = 0; q < 16; ++q) d += su[i * 160 + j * 16 + q] * sv[j * 16 + q];
            sb[idx] += d;
        }
        __syncthreads();
        if (t < 64) {
            float mx = sb[t * 10];
            for (int j = 1; j < 10; ++j) mx = fmaxf(mx, sb[t * 10 + j]);
            float sum = 0.0f;
            for (int j = 0; j < 10; ++j) { const float e = __expf(sb[t * 10 + j] - mx); sc[t * 10 + j] = e; sum += e; }
            const float inv = 1.0f / sum;
            for (int j = 0; j < 10; ++j) sc[t * 10 + j] *= inv;
        }
        __syncthreads();
        if (t < 160) {
            const int j = t >> 4;
            float s = 0.0f;
            for (int i = 0; i < 64; ++i) s += sc[i * 10 + j] * su[i * 160 + t];
            ss[t] = s;
        }
        __syncthreads();
        if (t < 10) {
            float l2 = 0.0f;
            for (int d = 0; d < 16; ++d) { const float x = ss[t * 16 + d]; l2 += x * x; }
            sscale[t] = l2 / (1.0f + l2) * __frsqrt_rn(l2);
        }
        __syncthreads();
        if (t < 160) sv[t] = ss[t] * sscale[t >> 4];
        __syncthreads();
    }
    if (t < 10) {
        float l2 = 0.0f;
        for (int d = 0; d < 16; ++d) { const float x = sv[t * 16 + d]; l2 += x * x; }
        const float p = sqrtf(l2);
        sscale[t] = p;
        probs_out[(long)n * 10 + t] = p;
    }
    __syncthreads();
    if (t == 0) {
        int best = 0; float bv = sscale[0];
        for (int j = 1; j < 10; ++j) if (sscale[j] > bv) { bv = sscale[j]; best = j; }
        slabel = best;
    }
    __syncthreads();
    if (t < 10) onehot[(long)n * 10 + t] = (t == slabel) ? 1.0f : 0.0f;
}

// y (N,170,8,8): c<10 -> one_hot broadcast; else u.reshape(N,8,8,160) transposed
__global__ void k_assemble(const float* __restrict__ u, const float* __restrict__ onehot,
                           float* __restrict__ y)
{
    const int n = blockIdx.x;
    for (int idx = threadIdx.x; idx < 170 * 64; idx += 256) {
        const int c = idx >> 6, hw = idx & 63;
        float v;
        if (c < 10) v = onehot[(long)n * 10 + c];
        else        v = u[(((long)n * 64 + hw) * 160) + (c - 10)];
        y[(long)n * 10880 + idx] = v;
    }
}

extern "C" void kernel_launch(void* const* d_in, const int* in_sizes, int n_in,
                              void* d_out, int out_size, void* d_ws, size_t ws_size,
                              hipStream_t stream) {
    (void)in_sizes; (void)n_in; (void)out_size; (void)ws_size;
    const float* input    = (const float*)d_in[0];
    const float* conv1_w  = (const float*)d_in[1];
    const float* conv1_b  = (const float*)d_in[2];
    const float* prim_w   = (const float*)d_in[3];
    const float* prim_b   = (const float*)d_in[4];
    const float* caps_w   = (const float*)d_in[5];
    const float* b_logits = (const float*)d_in[6];

    char* ws = (char*)d_ws;
    size_t off = 0;
    auto alloc = [&](size_t bytes) -> void* {
        void* p = ws + off; off += (bytes + 255) & ~(size_t)255; return p;
    };
    bf16_t* A1     = (bf16_t*)alloc(147456ull * 256 * 2);          // im2col conv1
    bf16_t* Wc1    = (bf16_t*)alloc(256ull * 256 * 2);             // conv1 weights bf16
    bf16_t* c1out  = (bf16_t*)alloc(147456ull * 256 * 2);          // conv1 out NHWC bf16
    bf16_t* Wp     = (bf16_t*)alloc(256ull * 20736 * 2);           // prim weights bf16
    float*  xpart  = (float*)alloc((size_t)SPLITK * 16384 * 256 * 4); // split-K partials
    float*  xprim  = (float*)alloc(16384ull * 256 * 4);            // prim conv out f32
    bf16_t* xsq    = (bf16_t*)alloc(16384ull * 256 * 2);           // squashed caps bf16
    bf16_t* Wc     = (bf16_t*)alloc(64ull * 160 * 256 * 2);        // caps weights bf16
    float*  uu     = (float*)alloc(256ull * 64 * 160 * 4);         // u predictions f32
    float*  onehot = (float*)alloc(256ull * 10 * 4);

    float* y     = (float*)d_out;
    float* probs = y + 256ull * 170 * 64;

    k_wc1    <<<256,           256, 0, stream>>>(conv1_w, Wc1);
    k_im2col1<<<147456,        256, 0, stream>>>(input, A1);
    // conv1: M=147456, N=256, K=256 (padded), fused bias+ReLU, bf16 NHWC out
    gemm_wmma<0, false, true, true, true><<<dim3(1152, 2, 1), 256, 0, stream>>>(
        A1, Wc1, c1out, conv1_b, 256, 256, 256, 256, 256, 0, 0, 0, 0);
    k_wp     <<<dim3(81, 256), 256, 0, stream>>>(prim_w, Wp);
    // prim conv (dominant 174 GFLOP): implicit im2col, M=16384, N=256,
    // K=20736 split 3-way (grid.z) -> 768 workgroups; partials reduced below
    gemm_wmma<1, false, false, false, false><<<dim3(128, 2, SPLITK), 256, 0, stream>>>(
        c1out, Wp, xpart, nullptr, 256, 20736 / SPLITK, 0, 20736, 256,
        0, 0, 16384L * 256, 20736 / SPLITK);
    k_reduce <<<16384,         256, 0, stream>>>(xpart, prim_b, xprim);
    k_squash <<<2048,          256, 0, stream>>>(xprim, xsq);
    k_wcaps  <<<dim3(160, 64), 256, 0, stream>>>(caps_w, Wc);
    // caps einsum: 64 batched GEMMs, M=256, N=160, K=256
    gemm_wmma<0, true, false, false, false><<<dim3(2, 2, 64), 256, 0, stream>>>(
        xsq, Wc, uu, nullptr, 160, 256, 16384, 256, 10240,
        256, (long)160 * 256, 160, 0);
    k_routing<<<256,           256, 0, stream>>>(uu, b_logits, probs, onehot);
    k_assemble<<<256,          256, 0, stream>>>(uu, onehot, y);
}
